// transformer_layer_19318762897745
// MI455X (gfx1250) — compile-verified, matrix operand order checked by reading, not run on hardware
//
#include <hip/hip_runtime.h>

// ---- problem constants (match setup_inputs) ----
#define Bk   2
#define Nk   2048
#define Ck   32
#define Dk   32
#define Kk   36          // knn_num
#define PRE  35          // C+3
#define PRE2 70          // 2*(C+3)
#define KPAD 96          // K padded to 3 x 32 for bf16 WMMA
#define MPAD 48          // 36 edges padded to 3 x 16
#define NCOL 1024        // D*D output columns of W_r
#define NPTS (Bk*Nk)

typedef __attribute__((ext_vector_type(16))) __bf16 v16bf;
typedef __attribute__((ext_vector_type(8)))  float  v8f;

__device__ __forceinline__ unsigned short f2bf_bits(float f) {
    union { float f; unsigned u; } a; a.f = f;
    return (unsigned short)((a.u + 0x7FFFu + ((a.u >> 16) & 1u)) >> 16);
}

// ---------------------------------------------------------------------------
// KNN: one block per query point. dist row in LDS, 36x argmin with the same
// tie-break as jax.lax.top_k (smaller index wins on equal distance).
// ---------------------------------------------------------------------------
__global__ __launch_bounds__(256) void knn_kernel(const float* __restrict__ xyz,
                                                  int* __restrict__ idx_out) {
    __shared__ float dist[Nk];
    __shared__ float sval[256];
    __shared__ int   sidx[256];
    const int p = blockIdx.x;          // global point
    const int b = p / Nk;
    const int t = threadIdx.x;

    const float qx = xyz[p * 3 + 0], qy = xyz[p * 3 + 1], qz = xyz[p * 3 + 2];
    const float q2 = qx * qx + qy * qy + qz * qz;
    for (int m = t; m < Nk; m += 256) {
        const int gp = b * Nk + m;
        const float x = xyz[gp * 3 + 0], y = xyz[gp * 3 + 1], z = xyz[gp * 3 + 2];
        const float m2 = x * x + y * y + z * z;
        dist[m] = q2 + m2 - 2.0f * (qx * x + qy * y + qz * z);  // mirrors reference formula
    }
    __syncthreads();

    for (int it = 0; it < Kk; ++it) {
        float bv = 3.4e38f; int bi = 0x7fffffff;
        for (int m = t; m < Nk; m += 256) {
            const float dv = dist[m];
            if (dv < bv || (dv == bv && m < bi)) { bv = dv; bi = m; }
        }
        sval[t] = bv; sidx[t] = bi;
        __syncthreads();
        for (int s = 128; s > 0; s >>= 1) {
            if (t < s) {
                const float ov = sval[t + s]; const int oi = sidx[t + s];
                if (ov < sval[t] || (ov == sval[t] && oi < sidx[t])) { sval[t] = ov; sidx[t] = oi; }
            }
            __syncthreads();
        }
        if (t == 0) {
            idx_out[p * Kk + it] = sidx[0];   // batch-local index, like reference
            dist[sidx[0]] = 3.4e38f;
        }
        __syncthreads();
    }
}

// ---------------------------------------------------------------------------
// v = relu(pre @ W_v + b_v) for every point (tiny).
// ---------------------------------------------------------------------------
__global__ __launch_bounds__(256) void v_kernel(const float* __restrict__ feature,
                                                const float* __restrict__ xyz,
                                                const float* __restrict__ W_v,
                                                const float* __restrict__ b_v,
                                                float* __restrict__ v_out) {
    const int gid = blockIdx.x * 256 + threadIdx.x;
    if (gid >= NPTS * Dk) return;
    const int p = gid >> 5, d = gid & 31;
    float acc = b_v[d];
    #pragma unroll 4
    for (int c = 0; c < Ck; ++c) acc += feature[p * Ck + c] * W_v[c * Dk + d];
    for (int c = 0; c < 3; ++c)  acc += xyz[p * 3 + c] * W_v[(Ck + c) * Dk + d];
    v_out[gid] = fmaxf(acc, 0.0f);
}

// ---------------------------------------------------------------------------
// Repack W_r (f32 [70,1024]) -> bf16 WMMA B-fragments, K padded to 96.
// Layout: [kTile(3)][nTile(64)][lane(32)][j(16)] so each lane's fragment is a
// contiguous 32-byte run (two uint4 loads, fully coalesced, L2-resident).
// ---------------------------------------------------------------------------
__global__ __launch_bounds__(256) void wr_repack_kernel(const float* __restrict__ W_r,
                                                        unsigned short* __restrict__ wr_frag) {
    const int gid = blockIdx.x * 256 + threadIdx.x;   // 3*64*32*16 = 98304
    if (gid >= 3 * 64 * 32 * 16) return;
    const int j    = gid & 15;
    const int lane = (gid >> 4) & 31;
    const int nt   = (gid >> 9) & 63;
    const int kt   = gid >> 15;
    const int n  = nt * 16 + (lane & 15);
    const int kk = ((j < 8) ? 0 : 16) + ((lane & 16) ? 8 : 0) + (j & 7);
    const int k  = kt * 32 + kk;
    const float v = (k < PRE2) ? W_r[k * NCOL + n] : 0.0f;
    wr_frag[gid] = f2bf_bits(v);
}

// ---------------------------------------------------------------------------
// Fused main kernel: one block (8 wave32) per point.
//   G[48x96] bf16 in LDS -> 9 A-fragments hoisted into registers (invariant
//   across column chunks) -> WMMA (bf16 16x16x32, f32 acc) streaming B from
//   L2 -> wbuf[48x128] per 128-col chunk -> per-(k,d) L1 scale ->
//   contraction with gv -> W_s epilogue.
// ---------------------------------------------------------------------------
__global__ __launch_bounds__(256) void fused_kernel(const float* __restrict__ feature,
                                                    const float* __restrict__ xyz,
                                                    const int*   __restrict__ idx,
                                                    const float* __restrict__ v_all,
                                                    const unsigned short* __restrict__ wr_frag,
                                                    const float* __restrict__ b_r,
                                                    const float* __restrict__ W_s,
                                                    const float* __restrict__ b_s,
                                                    float* __restrict__ out,
                                                    int out_size) {
    __shared__ float          pre_n[PRE];
    __shared__ int            idx_l[Kk];
    __shared__ float          nbr[Kk][PRE];
    __shared__ unsigned short gA[MPAD * KPAD];     // bf16 bits, row-major
    __shared__ float          gv[Kk][Dk];
    __shared__ float          wbuf[MPAD][128];     // one 128-col chunk of weights
    __shared__ float          scale_l[Kk * 4];     // per-(k, d_local) L1 scale
    __shared__ float          outpart[8][Dk];
    __shared__ float          outsum[Dk];

    const int p = blockIdx.x;
    const int b = p / Nk;
    const int t = threadIdx.x;
    const int lane = t & 31;
    const int w    = t >> 5;

    if (t < PRE) pre_n[t] = (t < Ck) ? feature[p * Ck + t] : xyz[p * 3 + (t - Ck)];
    if (t < Kk)  idx_l[t] = idx[p * Kk + t];
    __syncthreads();

    // gather neighbor pre rows and gv
    for (int e = t; e < Kk * PRE; e += 256) {
        const int k = e / PRE, c = e % PRE;
        const int nb = b * Nk + idx_l[k];
        nbr[k][c] = (c < Ck) ? feature[nb * Ck + c] : xyz[nb * 3 + (c - Ck)];
    }
    for (int e = t; e < Kk * Dk; e += 256) {
        const int k = e >> 5, d = e & 31;
        gv[k][d] = v_all[(b * Nk + idx_l[k]) * Dk + d];
    }
    __syncthreads();

    // build G = [gather - pre | pre] padded to 48x96, bf16
    for (int e = t; e < MPAD * KPAD; e += 256) {
        const int r = e / KPAD, c = e % KPAD;
        float val = 0.0f;
        if (r < Kk) {
            if (c < PRE)       val = nbr[r][c] - pre_n[c];
            else if (c < PRE2) val = pre_n[c - PRE];
        }
        gA[e] = f2bf_bits(val);
    }
    __syncthreads();

    // ---- hoist all 9 A fragments into registers: invariant over chunks ----
    union AF { v16bf v; unsigned u[8]; };
    AF Afrag[9];
    {
        const unsigned* gA32 = (const unsigned*)gA;
        #pragma unroll
        for (int mt = 0; mt < 3; ++mt) {
            const int row = mt * 16 + (lane & 15);
            #pragma unroll
            for (int kt = 0; kt < 3; ++kt) {
                const int kb = kt * 32 + ((lane & 16) ? 8 : 0);
                #pragma unroll
                for (int pq = 0; pq < 8; ++pq) {
                    const int k = kb + ((pq < 4) ? 0 : 16) + ((pq & 3) * 2);  // contiguous bf16 pairs
                    Afrag[mt * 3 + kt].u[pq] = gA32[(row * KPAD + k) >> 1];
                }
            }
        }
    }

    float accO = 0.0f;   // this thread's partial of out[e = lane], group w

    for (int ch = 0; ch < 8; ++ch) {
        // ---- WMMA phase: wave w owns N-tile nt (16 columns of this chunk) ----
        const int nt = ch * 8 + w;
        const float bias = b_r[nt * 16 + (lane & 15)];
        // B fragments for the 3 k-steps of this wave's N-tile (L2-resident stream)
        union BF { v16bf v; uint4 q[2]; };
        BF Bf[3];
        #pragma unroll
        for (int kt = 0; kt < 3; ++kt) {
            const uint4* bp = (const uint4*)(wr_frag + (((kt * 64 + nt) * 32 + lane) << 4));
            Bf[kt].q[0] = bp[0]; Bf[kt].q[1] = bp[1];
        }
        #pragma unroll
        for (int mt = 0; mt < 3; ++mt) {
            v8f acc = {};
            #pragma unroll
            for (int kt = 0; kt < 3; ++kt) {
                acc = __builtin_amdgcn_wmma_f32_16x16x32_bf16(
                        /*neg_a=*/false, Afrag[mt * 3 + kt].v, /*neg_b=*/false, Bf[kt].v,
                        /*c_mod=*/(short)0, acc, /*reuse_a=*/false, /*reuse_b=*/false);
            }
            // spill C tile (+ b_r bias) into the chunk buffer
            const int rbase = mt * 16 + ((lane & 16) ? 8 : 0);
            const int col   = w * 16 + (lane & 15);
            #pragma unroll
            for (int r = 0; r < 8; ++r) wbuf[rbase + r][col] = acc[r] + bias;
        }
        __syncthreads();

        // ---- per-(k, d_local) L1 scale: sqrt(32) / sum_e(|w| + 1e-7) ----
        if (t < Kk * 4) {
            const int k = t >> 2, dl = t & 3;
            float s = 32.0f * 1e-7f;
            #pragma unroll
            for (int e = 0; e < 32; ++e) s += fabsf(wbuf[k][dl * 32 + e]);
            scale_l[t] = 5.6568542494923806f / s;
        }
        __syncthreads();

        // ---- contraction: out[e] += gv[k,d] * scale * w[k,d,e] ----
        #pragma unroll
        for (int i = 0; i < 18; ++i) {
            const int pp = w + 8 * i;            // 144 (k,dl) pairs split over 8 waves
            const int k = pp >> 2, dl = pp & 3;
            accO += gv[k][ch * 4 + dl] * scale_l[pp] * wbuf[k][dl * 32 + lane];
        }
        __syncthreads();   // wbuf reused next chunk
    }

    outpart[w][lane] = accO;
    __syncthreads();
    if (t < Dk) {
        float s = 0.0f;
        #pragma unroll
        for (int g = 0; g < 8; ++g) s += outpart[g][t];
        outsum[t] = s;
    }
    __syncthreads();
    if (t < Dk) {
        float o = b_s[t];
        #pragma unroll 4
        for (int e = 0; e < Dk; ++e) o += outsum[e] * W_s[e * Dk + t];
        out[p * Dk + t] = o;
    }
    // tuple's second element: N
    if (p == 0 && t == 0 && out_size > NPTS * Dk) out[NPTS * Dk] = (float)Nk;
}

// ---------------------------------------------------------------------------
extern "C" void kernel_launch(void* const* d_in, const int* in_sizes, int n_in,
                              void* d_out, int out_size, void* d_ws, size_t ws_size,
                              hipStream_t stream) {
    const float* feature = (const float*)d_in[0];
    const float* xyz     = (const float*)d_in[1];
    const float* W_r     = (const float*)d_in[2];
    const float* b_r     = (const float*)d_in[3];
    const float* W_v     = (const float*)d_in[4];
    const float* b_v     = (const float*)d_in[5];
    const float* W_s     = (const float*)d_in[6];
    const float* b_s     = (const float*)d_in[7];
    (void)in_sizes; (void)n_in; (void)ws_size;

    char* ws = (char*)d_ws;
    int*            idx_ws  = (int*)ws;                                  // 4096*36*4   = 589824 B
    float*          v_ws    = (float*)(ws + 589824);                     // 4096*32*4   = 524288 B
    unsigned short* wr_frag = (unsigned short*)(ws + 589824 + 524288);   // 98304*2     = 196608 B
    float* out = (float*)d_out;

    knn_kernel<<<NPTS, 256, 0, stream>>>(xyz, idx_ws);
    v_kernel<<<(NPTS * Dk + 255) / 256, 256, 0, stream>>>(feature, xyz, W_v, b_v, v_ws);
    wr_repack_kernel<<<(3 * 64 * 32 * 16 + 255) / 256, 256, 0, stream>>>(W_r, wr_frag);
    fused_kernel<<<NPTS, 256, 0, stream>>>(feature, xyz, idx_ws, v_ws, wr_frag,
                                           b_r, W_s, b_s, out, out_size);
}
